// MultiHeadAttention_62122406969480
// MI455X (gfx1250) — matrix operand order repaired and checked by reading, not compiled, not run on hardware
//
#include <hip/hip_runtime.h>

// ---------------------------------------------------------------------------
// MultiHeadAttention forward for MI455X (gfx1250), bf16 WMMA + fused softmax,
// async global->LDS tile staging (ASYNCcnt path) where available.
// B=2, S=2048, D=1024, H=16, h=64.  Output = [x (B,S,D) fp32 | attn (B,H,S,S) fp32]
// ---------------------------------------------------------------------------

#define S_LEN   2048
#define DMODEL  1024
#define NHEADS  16
#define HDIM    64
#define BATCH   2
#define MROWS   (BATCH * S_LEN)   // 4096

typedef __bf16 bf16_t;
typedef __attribute__((ext_vector_type(16))) __bf16 v16bf;
typedef __attribute__((ext_vector_type(8)))  __bf16 v8bf;
typedef __attribute__((ext_vector_type(8)))  float  v8f;
typedef __attribute__((ext_vector_type(4)))  int    v4i;
typedef __attribute__((address_space(3)))    v4i    lds_v4i;  // LDS-qualified

__device__ __forceinline__ v8f zero_v8f() {
  v8f z = {0.f, 0.f, 0.f, 0.f, 0.f, 0.f, 0.f, 0.f};
  return z;
}

__device__ __forceinline__ v8f wmma_bf16(v16bf a, v16bf b, v8f c) {
  // (neg_a, A, neg_b, B, c_mod, C, reuse_a, reuse_b)
  return __builtin_amdgcn_wmma_f32_16x16x32_bf16(false, a, false, b, (short)0, c,
                                                 false, false);
}

// ---- async global -> LDS 16B copy (GLOBAL_LOAD_ASYNC_TO_LDS_B128) ----------
#if __has_builtin(__builtin_amdgcn_global_load_async_to_lds_b128)
#define HAVE_ASYNC_LDS 1
#else
#define HAVE_ASYNC_LDS 0
#endif

__device__ __forceinline__ void async_b128(void* lds_dst, const void* gsrc) {
#if HAVE_ASYNC_LDS
  // prototype (from clang diagnostic): param 0 is `int __vector(4) *`
  __builtin_amdgcn_global_load_async_to_lds_b128((v4i*)gsrc, (lds_v4i*)lds_dst,
                                                 0, 0);
#else
  *(v8bf*)lds_dst = *(const v8bf*)gsrc;
#endif
}

__device__ __forceinline__ void async_wait0() {
#if HAVE_ASYNC_LDS
#if __has_builtin(__builtin_amdgcn_s_wait_asynccnt)
  __builtin_amdgcn_s_wait_asynccnt(0);
#else
  asm volatile("s_wait_asynccnt 0x0" ::: "memory");
#endif
#endif
}

// ---- WMMA fragment builders ------------------------------------------------
// A fragment (16xK, 16-bit): lane l<16 holds row M=l, K = h*16..+15,
// h = lane>>4 (contiguous k order used here, not the ISA table order; B matches).
__device__ __forceinline__ v16bf load_a_frag(const bf16_t* base, int ldk, int lane) {
  const int l = lane & 15, h = lane >> 4;
  const v8bf lo = *(const v8bf*)(base + l * ldk + h * 16);
  const v8bf hi = *(const v8bf*)(base + l * ldk + h * 16 + 8);
  v16bf a;
#pragma unroll
  for (int i = 0; i < 8; ++i) { a[i] = lo[i]; a[i + 8] = hi[i]; }
  return a;
}

// B fragment (Kx16) from an [n][k] layout: lane n<16 holds col N=n, K=h*16..+15.
__device__ __forceinline__ v16bf load_b_frag(const bf16_t* base, int ldk, int lane) {
  const int n = lane & 15, h = lane >> 4;
  const v8bf lo = *(const v8bf*)(base + n * ldk + h * 16);
  const v8bf hi = *(const v8bf*)(base + n * ldk + h * 16 + 8);
  v16bf b;
#pragma unroll
  for (int i = 0; i < 8; ++i) { b[i] = lo[i]; b[i + 8] = hi[i]; }
  return b;
}

// Same fragments, but source tile staged in LDS as raw fp32 (async copies can't
// convert); fp32 -> bf16 conversion happens here in registers (v_cvt_pk_bf16).
__device__ __forceinline__ v16bf load_a_frag_f32(const float* base, int ldk, int lane) {
  const int l = lane & 15, h = lane >> 4;
  const float* p0 = base + l * ldk + h * 16;
  const float* p1 = p0 + 8;
  v16bf a;
#pragma unroll
  for (int i = 0; i < 8; ++i) { a[i] = (bf16_t)p0[i]; a[i + 8] = (bf16_t)p1[i]; }
  return a;
}

__device__ __forceinline__ v16bf load_b_frag_f32(const float* base, int ldk, int lane) {
  const int n = lane & 15, h = lane >> 4;
  const float* p = base + n * ldk + h * 16;
  v16bf b;
#pragma unroll
  for (int i = 0; i < 16; ++i) b[i] = (bf16_t)p[i];
  return b;
}

// ---------------------------------------------------------------------------
// Kernel 1: projection GEMM  out = X @ W^T + bias  (fp32 in, bf16 out)
// MODE 0: head-split [B,H,S,h] (q, k);  MODE 1: transposed [B,H,h,S] (v).
// Block 128 thr (4 waves); BM=64, BN=128, BK=32; wave w owns M-tile row w
// (8 WMMA per wave per k-step per staged A-fragment).
// ---------------------------------------------------------------------------
template <int MODE>
__global__ void __launch_bounds__(128) proj_qkv_kernel(
    const float* __restrict__ X, const float* __restrict__ W,
    const float* __restrict__ bias, bf16_t* __restrict__ out) {
  constexpr int BM = 64, BN = 128, BK = 32;
  extern __shared__ char smem[];
  float* Af = (float*)smem;        // [BM][BK] fp32 (8 KB)
  float* Bf = Af + BM * BK;        // [BN][BK] fp32 (16 KB), W is [n][k] already

  const int tid = threadIdx.x;
  const int lane = tid & 31;
  const int wave = tid >> 5;
  const int m0 = blockIdx.x * BM;
  const int n0 = blockIdx.y * BN;

  v8f acc[8];
#pragma unroll
  for (int i = 0; i < 8; ++i) acc[i] = zero_v8f();

  for (int kt = 0; kt < DMODEL; kt += BK) {
    // A tile: 64x32 fp32 = 512 x b128, 4 per thread (async, no VGPR staging)
#pragma unroll
    for (int i = 0; i < 4; ++i) {
      const int f = tid + i * 128;     // float4 index; 8 per row
      const int r = f >> 3;
      const int c = (f & 7) * 4;
      async_b128(Af + r * BK + c, X + (size_t)(m0 + r) * DMODEL + kt + c);
    }
    // B tile: 128x32 fp32 = 1024 x b128, 8 per thread
#pragma unroll
    for (int i = 0; i < 8; ++i) {
      const int f = tid + i * 128;
      const int r = f >> 3;
      const int c = (f & 7) * 4;
      async_b128(Bf + r * BK + c, W + (size_t)(n0 + r) * DMODEL + kt + c);
    }
    if (kt + BK < DMODEL) {  // global_prefetch_b8 of next K tile
      __builtin_prefetch(X + (size_t)(m0 + (tid >> 3)) * DMODEL + kt + BK, 0, 0);
      __builtin_prefetch(W + (size_t)(n0 + (tid >> 3)) * DMODEL + kt + BK, 0, 0);
    }
    async_wait0();
    __syncthreads();

    const v16bf a = load_a_frag_f32(Af + wave * 16 * BK, BK, lane);
#pragma unroll
    for (int nt = 0; nt < 8; ++nt) {
      const v16bf b = load_b_frag_f32(Bf + nt * 16 * BK, BK, lane);
      acc[nt] = wmma_bf16(a, b, acc[nt]);
    }
    __syncthreads();
  }

  // epilogue: C layout = lane%16 -> N, vgpr r -> M = r + 8*(lane>>4)
  const int rbase = m0 + wave * 16 + ((lane >> 4) << 3);
#pragma unroll
  for (int nt = 0; nt < 8; ++nt) {
    const int col = n0 + nt * 16 + (lane & 15);
    const float bv = bias[col];
    const int hh = col >> 6;   // head index
    const int dd = col & 63;   // within-head dim
#pragma unroll
    for (int r = 0; r < 8; ++r) {
      const int m = rbase + r;
      const int bb = m >> 11;          // batch
      const int ss = m & (S_LEN - 1);
      const float v = acc[nt][r] + bv;
      size_t idx;
      if (MODE == 0)
        idx = ((size_t)((bb * NHEADS + hh) * S_LEN + ss)) * HDIM + dd;
      else
        idx = ((size_t)((bb * NHEADS + hh) * HDIM + dd)) * S_LEN + ss;
      out[idx] = (bf16_t)v;
    }
  }
}

// ---------------------------------------------------------------------------
// Kernel 2: fused attention per (b, h, 16 q-rows).
//  phase 1: E[16][2048] = (Qtile @ K^T) * 0.125, masked   (WMMA, fp32 in LDS)
//  phase 2: row softmax in LDS; one coalesced fp32 write of the attention map;
//           bf16 probabilities P kept in LDS
//  phase 3: ctx_tile = P @ V  (WMMA over K=2048, V pre-transposed [B,H,h,S])
// LDS: 128KB E + 64KB P + 2KB Q + scratch  (fits in 320KB/WGP)
// ---------------------------------------------------------------------------
__global__ void __launch_bounds__(128) attn_kernel(
    const bf16_t* __restrict__ qh, const bf16_t* __restrict__ kh,
    const bf16_t* __restrict__ vT, const int* __restrict__ mask,
    float* __restrict__ attn_out, bf16_t* __restrict__ ctx) {
  extern __shared__ char smem[];
  float*  E      = (float*)smem;                  // 16 * 2048 fp32
  bf16_t* P      = (bf16_t*)(E + 16 * S_LEN);     // 16 * 2048 bf16
  bf16_t* Qs     = P + 16 * S_LEN;                // 16 * 64 bf16
  float*  red    = (float*)(Qs + 16 * HDIM);      // 128
  float*  rowmax = red + 128;                     // 16
  float*  rowinv = rowmax + 16;                   // 16

  const int tid = threadIdx.x, lane = tid & 31, wave = tid >> 5;
  const int bh = blockIdx.y, b = bh >> 4, hh = bh & 15;
  const int q0 = blockIdx.x * 16;

  {  // stage the 16x64 Q tile asynchronously (one b128 per thread)
    const bf16_t* Qg = qh + ((size_t)bh * S_LEN + q0) * HDIM;
    const int r = tid >> 3, c = (tid & 7) * 8;
    async_b128(Qs + r * HDIM + c, Qg + (size_t)r * HDIM + c);
  }
  async_wait0();
  __syncthreads();

  const v16bf qa0 = load_a_frag(Qs, HDIM, lane);       // d = 0..31
  const v16bf qa1 = load_a_frag(Qs + 32, HDIM, lane);  // d = 32..63

  // ---- phase 1: energy stripe; wave w handles column tiles w, w+4, ...
  for (int ct = wave; ct < S_LEN / 16; ct += 4) {
    const int s0 = ct * 16;
    const bf16_t* Kb = kh + ((size_t)bh * S_LEN + s0) * HDIM;  // [s][d] == [n][k]
    const v16bf b0 = load_b_frag(Kb, HDIM, lane);
    const v16bf b1 = load_b_frag(Kb + 32, HDIM, lane);
    v8f e = zero_v8f();
    e = wmma_bf16(qa0, b0, e);
    e = wmma_bf16(qa1, b1, e);

    const int colg = s0 + (lane & 15);
    const bool live = (mask[b * S_LEN + colg] != 0);
    const int mrow = (lane >> 4) << 3;
#pragma unroll
    for (int r = 0; r < 8; ++r) {
      float ev = e[r] * 0.125f;           // 1/sqrt(64)
      if (!live) ev = -1e10f;             // NEG_INF
      E[(mrow + r) * S_LEN + colg] = ev;
    }
  }
  __syncthreads();

  // ---- phase 2: softmax (8 threads per row, 256 elems each)
  const int rr = tid >> 3;
  const int seg = (tid & 7) * (S_LEN / 8);
  float mx = -3.4e38f;
  for (int j = 0; j < S_LEN / 8; ++j) mx = fmaxf(mx, E[rr * S_LEN + seg + j]);
  red[tid] = mx;
  __syncthreads();
  if (tid < 16) {
    float m = red[tid * 8];
#pragma unroll
    for (int k = 1; k < 8; ++k) m = fmaxf(m, red[tid * 8 + k]);
    rowmax[tid] = m;
  }
  __syncthreads();
  const float rm = rowmax[rr];
  float sum = 0.f;
  for (int j = 0; j < S_LEN / 8; ++j) {
    const int idx = rr * S_LEN + seg + j;
    const float pe = __expf(E[idx] - rm);
    E[idx] = pe;
    sum += pe;
  }
  red[tid] = sum;
  __syncthreads();
  if (tid < 16) {
    float s = 0.f;
#pragma unroll
    for (int k = 0; k < 8; ++k) s += red[tid * 8 + k];
    rowinv[tid] = 1.0f / s;
  }
  __syncthreads();

  {  // normalize, write attention map (coalesced), build bf16 P
    float* ob = attn_out + ((size_t)bh * S_LEN + q0) * S_LEN;
    for (int idx = tid; idx < 16 * S_LEN; idx += 128) {
      const int r = idx >> 11;  // idx / 2048
      const float p = E[idx] * rowinv[r];
      ob[(size_t)idx] = p;      // [r][c] flat == r*S_LEN + c
      P[idx] = (bf16_t)p;
    }
  }
  __syncthreads();

  // ---- phase 3: ctx tile = P @ V ; wave w owns output columns d = w*16..+15
  v8f o = zero_v8f();
  const bf16_t* Vb = vT + ((size_t)bh * HDIM + wave * 16) * S_LEN;  // [d][s]
  for (int k0 = 0; k0 < S_LEN; k0 += 32) {
    const v16bf pa = load_a_frag(P + k0, S_LEN, lane);
    const v16bf vb = load_b_frag(Vb + k0, S_LEN, lane);
    o = wmma_bf16(pa, vb, o);
  }
  {
    const int dcol = hh * HDIM + wave * 16 + (lane & 15);
    const int mrow = (lane >> 4) << 3;
#pragma unroll
    for (int r = 0; r < 8; ++r) {
      ctx[((size_t)(b * S_LEN + q0 + mrow + r)) * DMODEL + dcol] = (bf16_t)o[r];
    }
  }
}

// ---------------------------------------------------------------------------
// Kernel 3: output projection  x = ctx @ Wo^T + bo  (bf16 A, fp32 W, fp32 out)
// Same tiling as kernel 1 (BM=64, BN=128, BK=32).
// ---------------------------------------------------------------------------
__global__ void __launch_bounds__(128) out_proj_kernel(
    const bf16_t* __restrict__ Xc, const float* __restrict__ W,
    const float* __restrict__ bias, float* __restrict__ out) {
  constexpr int BM = 64, BN = 128, BK = 32;
  extern __shared__ char smem[];
  bf16_t* As = (bf16_t*)smem;              // [BM][BK] bf16 (4 KB)
  float*  Bf = (float*)(As + BM * BK);     // [BN][BK] fp32 (16 KB)

  const int tid = threadIdx.x;
  const int lane = tid & 31;
  const int wave = tid >> 5;
  const int m0 = blockIdx.x * BM;
  const int n0 = blockIdx.y * BN;

  v8f acc[8];
#pragma unroll
  for (int i = 0; i < 8; ++i) acc[i] = zero_v8f();

  for (int kt = 0; kt < DMODEL; kt += BK) {
    // A tile (bf16): 64x32 = 256 x b128, 2 per thread
#pragma unroll
    for (int i = 0; i < 2; ++i) {
      const int v = tid + i * 128;     // b128 index; 4 per row
      const int r = v >> 2;
      const int c = (v & 3) * 8;
      async_b128(As + r * BK + c, Xc + (size_t)(m0 + r) * DMODEL + kt + c);
    }
    // B tile (fp32): 128x32 = 1024 x b128, 8 per thread
#pragma unroll
    for (int i = 0; i < 8; ++i) {
      const int f = tid + i * 128;
      const int r = f >> 3;
      const int c = (f & 7) * 4;
      async_b128(Bf + r * BK + c, W + (size_t)(n0 + r) * DMODEL + kt + c);
    }
    async_wait0();
    __syncthreads();

    const v16bf a = load_a_frag(As + wave * 16 * BK, BK, lane);
#pragma unroll
    for (int nt = 0; nt < 8; ++nt) {
      const v16bf b = load_b_frag_f32(Bf + nt * 16 * BK, BK, lane);
      acc[nt] = wmma_bf16(a, b, acc[nt]);
    }
    __syncthreads();
  }

  const int rbase = m0 + wave * 16 + ((lane >> 4) << 3);
#pragma unroll
  for (int nt = 0; nt < 8; ++nt) {
    const int col = n0 + nt * 16 + (lane & 15);
    const float bv = bias[col];
#pragma unroll
    for (int r = 0; r < 8; ++r) {
      out[(size_t)(rbase + r) * DMODEL + col] = acc[nt][r] + bv;
    }
  }
}

// ---------------------------------------------------------------------------
extern "C" void kernel_launch(void* const* d_in, const int* in_sizes, int n_in,
                              void* d_out, int out_size, void* d_ws, size_t ws_size,
                              hipStream_t stream) {
  (void)in_sizes; (void)n_in; (void)out_size; (void)ws_size;

  const float* Q    = (const float*)d_in[0];
  const float* K    = (const float*)d_in[1];
  const float* V    = (const float*)d_in[2];
  const int*   mask = (const int*)d_in[3];
  const float* Wq   = (const float*)d_in[4];
  const float* bq   = (const float*)d_in[5];
  const float* Wk   = (const float*)d_in[6];
  const float* bk   = (const float*)d_in[7];
  const float* Wv   = (const float*)d_in[8];
  const float* bv   = (const float*)d_in[9];
  const float* Wo   = (const float*)d_in[10];
  const float* bo   = (const float*)d_in[11];

  const size_t head_elems = (size_t)BATCH * NHEADS * S_LEN * HDIM;  // 4.19M
  bf16_t* qh  = (bf16_t*)d_ws;
  bf16_t* kh  = qh + head_elems;
  bf16_t* vT  = kh + head_elems;
  bf16_t* ctx = vT + head_elems;   // [B, S, D] bf16

  float* x_out    = (float*)d_out;
  float* attn_out = x_out + (size_t)MROWS * DMODEL;  // tuple: (x, attention)

  const dim3 gp(MROWS / 64, DMODEL / 128);           // 64 x 8 blocks
  const size_t smem_p = (size_t)(64 * 32 + 128 * 32) * sizeof(float);   // 24 KB
  const size_t smem_o = (size_t)64 * 32 * sizeof(bf16_t)
                      + (size_t)128 * 32 * sizeof(float);               // 20 KB

  proj_qkv_kernel<0><<<gp, dim3(128), smem_p, stream>>>(Q, Wq, bq, qh);
  proj_qkv_kernel<0><<<gp, dim3(128), smem_p, stream>>>(K, Wk, bk, kh);
  proj_qkv_kernel<1><<<gp, dim3(128), smem_p, stream>>>(V, Wv, bv, vT);

  const dim3 ga(S_LEN / 16, BATCH * NHEADS);         // 128 x 32 blocks
  const size_t smem_a = (size_t)16 * S_LEN * sizeof(float)    // E
                      + (size_t)16 * S_LEN * sizeof(bf16_t)   // P
                      + (size_t)16 * HDIM  * sizeof(bf16_t)   // Qs
                      + (size_t)(128 + 16 + 16) * sizeof(float);
  attn_kernel<<<ga, dim3(128), smem_a, stream>>>(qh, kh, vT, mask, attn_out, ctx);

  out_proj_kernel<<<gp, dim3(128), smem_o, stream>>>(ctx, Wo, bo, x_out);
}
